// CRFasRNN_71287867179181
// MI455X (gfx1250) — matrix operand order repaired and checked by reading, not compiled, not run on hardware
//
#include <hip/hip_runtime.h>
#include <hip/hip_bf16.h>

#define Hn 512
#define Wn 512
#define HW (Hn * Wn)
#define Bn 2
#define Ln 21
#define Cn 3
#define Rr 20
#define NIT 5

typedef __attribute__((ext_vector_type(16))) __bf16 bf16x16;
typedef __attribute__((ext_vector_type(8)))  float  f32x8;

__device__ __forceinline__ float winCount(int i, int n) {
    int lo = (i - Rr > 0) ? (i - Rr) : 0;
    int hi = (i + Rr < n - 1) ? (i + Rr) : (n - 1);
    return (float)(hi - lo + 1);
}

// ---------------------------------------------------------------------------
// Q0 = softmax(-E0) over the label axis (one thread per pixel)
// ---------------------------------------------------------------------------
__global__ void softmax0_k(const float* __restrict__ E0, float* __restrict__ Q) {
    int p = blockIdx.x * blockDim.x + threadIdx.x;
    int b = p / HW, pix = p % HW;
    const float* e = E0 + (size_t)b * Ln * HW + pix;
    float v[Ln];
    float mx = -1e30f;
#pragma unroll
    for (int l = 0; l < Ln; ++l) { v[l] = -e[(size_t)l * HW]; mx = fmaxf(mx, v[l]); }
    float s = 0.0f;
#pragma unroll
    for (int l = 0; l < Ln; ++l) { v[l] = __expf(v[l] - mx); s += v[l]; }
    float inv = 1.0f / s;
    float* q = Q + (size_t)b * Ln * HW + pix;
#pragma unroll
    for (int l = 0; l < Ln; ++l) q[(size_t)l * HW] = v[l] * inv;
}

// ---------------------------------------------------------------------------
// Precompute pack: ch0..2 = I, ch3..8 = unique I*I products  (9 ch per batch)
// ---------------------------------------------------------------------------
__global__ void pack9_k(const float* __restrict__ I, float* __restrict__ P) {
    int p = blockIdx.x * blockDim.x + threadIdx.x;
    int b = p / HW, pix = p % HW;
    const float* ib = I + (size_t)b * Cn * HW + pix;
    float i0 = ib[0], i1 = ib[(size_t)HW], i2 = ib[(size_t)2 * HW];
    float* pb = P + (size_t)b * 9 * HW + pix;
    pb[0]               = i0;
    pb[(size_t)1 * HW]  = i1;
    pb[(size_t)2 * HW]  = i2;
    pb[(size_t)3 * HW]  = i0 * i0;
    pb[(size_t)4 * HW]  = i0 * i1;
    pb[(size_t)5 * HW]  = i0 * i2;
    pb[(size_t)6 * HW]  = i1 * i1;
    pb[(size_t)7 * HW]  = i1 * i2;
    pb[(size_t)8 * HW]  = i2 * i2;
}

// ---------------------------------------------------------------------------
// Horizontal box filter, in place. One 512-thread block per row.
// Row is brought into LDS with the CDNA5 async-to-LDS path (ASYNCcnt),
// then an LDS Hillis-Steele inclusive scan reproduces the reference
// cumsum windowing: cs[min(i+r,n-1)] - (i>=r+1 ? cs[i-r-1] : 0).
// ---------------------------------------------------------------------------
__global__ void hbox_k(float* __restrict__ F) {
    __shared__ float cs[Wn];
    size_t row = blockIdx.x;
    float* r = F + row * (size_t)Wn;
    int t = threadIdx.x;

    // per-lane async copy: LDS[&cs[t]] = MEM[&r[t]]
    unsigned lds_off = (unsigned)(unsigned long long)(&cs[t]);
    unsigned long long gaddr = (unsigned long long)(r + t);
    asm volatile("global_load_async_to_lds_b32 %0, %1, off"
                 :: "v"(lds_off), "v"(gaddr) : "memory");
    asm volatile("s_wait_asynccnt 0x0" ::: "memory");
    __syncthreads();

    for (int off = 1; off < Wn; off <<= 1) {
        float v = (t >= off) ? cs[t - off] : 0.0f;
        __syncthreads();
        cs[t] += v;
        __syncthreads();
    }
    int hiIdx = (t + Rr < Wn - 1) ? (t + Rr) : (Wn - 1);
    float hi = cs[hiIdx];
    float lo = (t >= Rr + 1) ? cs[t - Rr - 1] : 0.0f;
    r[t] = hi - lo;
}

// ---------------------------------------------------------------------------
// Vertical box filter: per-column sliding window, 4 columns/thread (b128).
// grid = (4 row-strips, nimages), block = 128 threads covering 512 columns.
// Each strip primes its own window sum, then slides. Out-of-place.
// ---------------------------------------------------------------------------
__global__ void vbox_k(const float* __restrict__ Fin, float* __restrict__ Fout) {
    int t = threadIdx.x;                       // 0..127 -> columns 4t..4t+3
    size_t img = blockIdx.y;
    const int rs = Wn / 4;                     // float4s per row
    int y0 = blockIdx.x * (Hn / 4);
    int y1 = y0 + (Hn / 4);
    const float4* in = (const float4*)(Fin + img * (size_t)HW) + t;
    float4* out = (float4*)(Fout + img * (size_t)HW) + t;

    float4 s = make_float4(0.f, 0.f, 0.f, 0.f);
    int lo = (y0 - Rr > 0) ? (y0 - Rr) : 0;
    int hi = (y0 + Rr < Hn - 1) ? (y0 + Rr) : (Hn - 1);
    for (int y = lo; y <= hi; ++y) {
        float4 v = in[(size_t)y * rs];
        s.x += v.x; s.y += v.y; s.z += v.z; s.w += v.w;
    }
    for (int y = y0; y < y1; ++y) {
        out[(size_t)y * rs] = s;
        int ya = y + Rr + 1;
        if (ya < Hn) {
            float4 v = in[(size_t)ya * rs];
            s.x += v.x; s.y += v.y; s.z += v.z; s.w += v.w;
            if (ya + 8 < Hn) __builtin_prefetch(in + (size_t)(ya + 8) * rs, 0, 1);
        }
        int yb = y - Rr;
        if (yb >= 0) {
            float4 v = in[(size_t)yb * rs];
            s.x -= v.x; s.y -= v.y; s.z -= v.z; s.w -= v.w;
        }
    }
}

// ---------------------------------------------------------------------------
// From boxed(I, I*I): mI = box(I)/N ; A = var_I + eps*Id ; invA (symmetric 3x3)
// ---------------------------------------------------------------------------
__global__ void inva_k(const float* __restrict__ BX, const float* __restrict__ eps,
                       float* __restrict__ mI, float* __restrict__ invA) {
    int p = blockIdx.x * blockDim.x + threadIdx.x;
    int b = p / HW, pix = p % HW;
    int x = pix % Wn, y = pix / Wn;
    float invN = 1.0f / (winCount(x, Wn) * winCount(y, Hn));
    float e = eps[0];
    const float* bb = BX + (size_t)b * 9 * HW + pix;
    float m0 = bb[0] * invN;
    float m1 = bb[(size_t)1 * HW] * invN;
    float m2 = bb[(size_t)2 * HW] * invN;
    float v00 = bb[(size_t)3 * HW] * invN - m0 * m0 + e;
    float v01 = bb[(size_t)4 * HW] * invN - m0 * m1;
    float v02 = bb[(size_t)5 * HW] * invN - m0 * m2;
    float v11 = bb[(size_t)6 * HW] * invN - m1 * m1 + e;
    float v12 = bb[(size_t)7 * HW] * invN - m1 * m2;
    float v22 = bb[(size_t)8 * HW] * invN - m2 * m2 + e;
    float c00 = v11 * v22 - v12 * v12;
    float c01 = v02 * v12 - v01 * v22;
    float c02 = v01 * v12 - v02 * v11;
    float c11 = v00 * v22 - v02 * v02;
    float c12 = v02 * v01 - v00 * v12;
    float c22 = v00 * v11 - v01 * v01;
    float invd = 1.0f / (v00 * c00 + v01 * c01 + v02 * c02);
    float* mo = mI + (size_t)b * 3 * HW + pix;
    mo[0] = m0; mo[(size_t)1 * HW] = m1; mo[(size_t)2 * HW] = m2;
    float* ao = invA + (size_t)b * 6 * HW + pix;
    ao[0]               = c00 * invd;
    ao[(size_t)1 * HW]  = c01 * invd;
    ao[(size_t)2 * HW]  = c02 * invd;
    ao[(size_t)3 * HW]  = c11 * invd;
    ao[(size_t)4 * HW]  = c12 * invd;
    ao[(size_t)5 * HW]  = c22 * invd;
}

// ---------------------------------------------------------------------------
// Fused label-mixing GEMM (bf16 WMMA) + guide products.
// P[b,0..20]     = MuQ = W_mu @ Q           (M,K padded 21->32)
// P[b,21+c*21+l] = I_c * MuQ_l              (84 channels total)
// One wave handles 16 pixels and both 16-row M tiles.
// ---------------------------------------------------------------------------
__global__ void __launch_bounds__(256) gemm_prod_k(const float* __restrict__ Wmu,
                                                   const float* __restrict__ Q,
                                                   const float* __restrict__ I,
                                                   float* __restrict__ P) {
    __shared__ float sW[Ln * Ln];
    int tid = threadIdx.x;
    for (int i = tid; i < Ln * Ln; i += 256) sW[i] = Wmu[i];
    __syncthreads();

    int wave = tid >> 5, lane = tid & 31;
    int tile = blockIdx.x * 8 + wave;           // 16-pixel tile id
    const int tilesPerB = HW / 16;
    int b = tile / tilesPerB;
    int pixBase = (tile % tilesPerB) * 16;

    // A operand (W_mu rows), CDNA5 16-bit A layout:
    // lanes 0-15: M=lane, K groups {0..7,16..23}; lanes 16-31: M=lane-16, +8
    int mlo = lane & 15;
    int kbase = (lane < 16) ? 0 : 8;
    bf16x16 a0, a1;
#pragma unroll
    for (int j = 0; j < 16; ++j) {
        int v = j >> 1;
        int k = ((v < 4) ? (2 * v) : (16 + 2 * (v - 4))) + kbase + (j & 1);
        float w0 = (k < Ln) ? sW[mlo * Ln + k] : 0.0f;
        int r1 = 16 + mlo;
        float w1 = (r1 < Ln && k < Ln) ? sW[r1 * Ln + k] : 0.0f;
        a0[j] = (__bf16)w0;
        a1[j] = (__bf16)w1;
    }

    // B operand (Q columns = pixels), CDNA5 16-bit B layout:
    // lane L: N = L%16; lanes 0-15 hold K {0..7,16..23}, lanes 16-31 hold K {8..15,24..31}
    const float* qb = Q + (size_t)b * Ln * HW + pixBase + (lane & 15);
    int kb = (lane < 16) ? 0 : 8;
    bf16x16 bt;
#pragma unroll
    for (int j = 0; j < 16; ++j) {
        int k = kb + j + ((j >= 8) ? 8 : 0);
        float qv = (k < Ln) ? qb[(size_t)k * HW] : 0.0f;
        bt[j] = (__bf16)qv;
    }

    f32x8 d0 = {};
    f32x8 d1 = {};
    d0 = __builtin_amdgcn_wmma_f32_16x16x32_bf16(false, a0, false, bt, (short)0, d0, false, false);
    d1 = __builtin_amdgcn_wmma_f32_16x16x32_bf16(false, a1, false, bt, (short)0, d1, false, false);

    // Epilogue: D layout -> lane holds pixel N=lane%16, rows M = vgpr (+8 hi half)
    int pix = pixBase + (lane & 15);
    const float* ib = I + (size_t)b * Cn * HW + pix;
    float i0 = ib[0], i1 = ib[(size_t)HW], i2 = ib[(size_t)2 * HW];
    float* pb = P + (size_t)b * 84 * HW + pix;
    int moff = (lane < 16) ? 0 : 8;
#pragma unroll
    for (int v = 0; v < 8; ++v) {
        int l0 = moff + v;                       // 0..15, always valid
        float q0 = d0[v];
        pb[(size_t)l0 * HW]        = q0;
        pb[(size_t)(21 + l0) * HW] = i0 * q0;
        pb[(size_t)(42 + l0) * HW] = i1 * q0;
        pb[(size_t)(63 + l0) * HW] = i2 * q0;
        int l1 = 16 + moff + v;                  // 16..31, valid only < 21
        if (l1 < Ln) {
            float q1 = d1[v];
            pb[(size_t)l1 * HW]        = q1;
            pb[(size_t)(21 + l1) * HW] = i0 * q1;
            pb[(size_t)(42 + l1) * HW] = i1 * q1;
            pb[(size_t)(63 + l1) * HW] = i2 * q1;
        }
    }
}

// ---------------------------------------------------------------------------
// Per-pixel guided-filter coefficients:
// mp, cov_Ip from boxed pack; a = invA * cov ; b = mp - a.mI
// Output pack: ch c*21+l = a_c,l ; ch 63+l = b_l (84 channels)
// ---------------------------------------------------------------------------
__global__ void ab_k(const float* __restrict__ BX, const float* __restrict__ mI,
                     const float* __restrict__ invA, float* __restrict__ P) {
    int p = blockIdx.x * blockDim.x + threadIdx.x;
    int b = p / HW, pix = p % HW;
    int x = pix % Wn, y = pix / Wn;
    float invN = 1.0f / (winCount(x, Wn) * winCount(y, Hn));
    const float* bx = BX + (size_t)b * 84 * HW + pix;
    const float* mi = mI + (size_t)b * 3 * HW + pix;
    const float* ia = invA + (size_t)b * 6 * HW + pix;
    float m0 = mi[0], m1 = mi[(size_t)HW], m2 = mi[(size_t)2 * HW];
    float A00 = ia[0];
    float A01 = ia[(size_t)1 * HW];
    float A02 = ia[(size_t)2 * HW];
    float A11 = ia[(size_t)3 * HW];
    float A12 = ia[(size_t)4 * HW];
    float A22 = ia[(size_t)5 * HW];
    float* pb = P + (size_t)b * 84 * HW + pix;
#pragma unroll
    for (int l = 0; l < Ln; ++l) {
        float mp = bx[(size_t)l * HW] * invN;
        float c0 = bx[(size_t)(21 + l) * HW] * invN - m0 * mp;
        float c1 = bx[(size_t)(42 + l) * HW] * invN - m1 * mp;
        float c2 = bx[(size_t)(63 + l) * HW] * invN - m2 * mp;
        float a0 = A00 * c0 + A01 * c1 + A02 * c2;
        float a1 = A01 * c0 + A11 * c1 + A12 * c2;
        float a2 = A02 * c0 + A12 * c1 + A22 * c2;
        float bl = mp - (a0 * m0 + a1 * m1 + a2 * m2);
        pb[(size_t)l * HW]        = a0;
        pb[(size_t)(21 + l) * HW] = a1;
        pb[(size_t)(42 + l) * HW] = a2;
        pb[(size_t)(63 + l) * HW] = bl;
    }
}

// ---------------------------------------------------------------------------
// q = sum_c box(a_c)/N * I_c + box(b)/N ; Q = softmax(-(E0 + q))
// ---------------------------------------------------------------------------
__global__ void final_k(const float* __restrict__ BX, const float* __restrict__ E0,
                        const float* __restrict__ I, float* __restrict__ Qout) {
    int p = blockIdx.x * blockDim.x + threadIdx.x;
    int b = p / HW, pix = p % HW;
    int x = pix % Wn, y = pix / Wn;
    float invN = 1.0f / (winCount(x, Wn) * winCount(y, Hn));
    const float* bx = BX + (size_t)b * 84 * HW + pix;
    const float* ib = I + (size_t)b * Cn * HW + pix;
    const float* e = E0 + (size_t)b * Ln * HW + pix;
    float i0 = ib[0], i1 = ib[(size_t)HW], i2 = ib[(size_t)2 * HW];
    float v[Ln];
    float mx = -1e30f;
#pragma unroll
    for (int l = 0; l < Ln; ++l) {
        float q = (bx[(size_t)l * HW] * i0 +
                   bx[(size_t)(21 + l) * HW] * i1 +
                   bx[(size_t)(42 + l) * HW] * i2 +
                   bx[(size_t)(63 + l) * HW]) * invN;
        float val = -(e[(size_t)l * HW] + q);
        v[l] = val;
        mx = fmaxf(mx, val);
    }
    float s = 0.0f;
#pragma unroll
    for (int l = 0; l < Ln; ++l) { v[l] = __expf(v[l] - mx); s += v[l]; }
    float inv = 1.0f / s;
    float* q = Qout + (size_t)b * Ln * HW + pix;
#pragma unroll
    for (int l = 0; l < Ln; ++l) q[(size_t)l * HW] = v[l] * inv;
}

// ---------------------------------------------------------------------------
extern "C" void kernel_launch(void* const* d_in, const int* in_sizes, int n_in,
                              void* d_out, int out_size, void* d_ws, size_t ws_size,
                              hipStream_t stream) {
    (void)in_sizes; (void)n_in; (void)out_size; (void)ws_size;
    const float* E0   = (const float*)d_in[0];   // [B,21,H,W]
    const float* Refs = (const float*)d_in[1];   // [B,3,H,W]
    const float* Wmu  = (const float*)d_in[2];   // [21,21]
    const float* eps  = (const float*)d_in[3];   // [1]

    float* ws = (float*)d_ws;
    float* Q    = ws;                                   // B*21*HW
    float* P    = Q    + (size_t)Bn * Ln * HW;          // B*84*HW (pack / ab)
    float* X    = P    + (size_t)Bn * 84 * HW;          // B*84*HW (boxed)
    float* mI   = X    + (size_t)Bn * 84 * HW;          // B*3*HW
    float* invA = mI   + (size_t)Bn * 3  * HW;          // B*6*HW

    dim3 pixGrid((Bn * HW) / 256), pixBlk(256);

    // ---- loop-invariant precompute (guide statistics + initial softmax) ----
    pack9_k<<<pixGrid, pixBlk, 0, stream>>>(Refs, P);
    hbox_k<<<Bn * 9 * Hn, Wn, 0, stream>>>(P);
    vbox_k<<<dim3(4, Bn * 9), 128, 0, stream>>>(P, X);
    inva_k<<<pixGrid, pixBlk, 0, stream>>>(X, eps, mI, invA);
    softmax0_k<<<pixGrid, pixBlk, 0, stream>>>(E0, Q);

    // ---- 5 mean-field iterations ----
    for (int it = 0; it < NIT; ++it) {
        gemm_prod_k<<<(Bn * HW / 16) / 8, 256, 0, stream>>>(Wmu, Q, Refs, P);
        hbox_k<<<Bn * 84 * Hn, Wn, 0, stream>>>(P);
        vbox_k<<<dim3(4, Bn * 84), 128, 0, stream>>>(P, X);
        ab_k<<<pixGrid, pixBlk, 0, stream>>>(X, mI, invA, P);
        hbox_k<<<Bn * 84 * Hn, Wn, 0, stream>>>(P);
        vbox_k<<<dim3(4, Bn * 84), 128, 0, stream>>>(P, X);
        float* qdst = (it == NIT - 1) ? (float*)d_out : Q;
        final_k<<<pixGrid, pixBlk, 0, stream>>>(X, E0, Refs, qdst);
    }
}
